// EuclideanCodebook_52209622450624
// MI455X (gfx1250) — matrix operand order repaired and checked by reading, not compiled, not run on hardware
//
#include <hip/hip_runtime.h>

typedef __attribute__((ext_vector_type(16))) __bf16 v16bf;
typedef __attribute__((ext_vector_type(8)))  __bf16 v8bf;
typedef __attribute__((ext_vector_type(8)))  float  v8f;
typedef __attribute__((ext_vector_type(4)))  float  v4f;

#define DIMS      256
#define KCODES    8192
#define CHUNK     32
#define THREADS   256
#define ROWS_FAST 128
#define NCHUNKS   (KCODES / CHUNK)

__device__ __forceinline__ unsigned short f32_bf16_rne(float f) {
  unsigned int u = __float_as_uint(f);
  u += 0x7FFFu + ((u >> 16) & 1u);
  return (unsigned short)(u >> 16);
}
__device__ __forceinline__ float bf16_f32(unsigned short h) {
  return __uint_as_float(((unsigned int)h) << 16);
}

union AB16 { v16bf v; v8bf h[2]; unsigned short us[16]; };

// ---------------------------------------------------------------------------
// Preprocess: split embed fp32 -> bf16 hi/lo (once), and |e|^2 per code.
// One wave per code: 32 lanes x 8 dims, coalesced 1KB reads / 512B writes.
// ---------------------------------------------------------------------------
__global__ __launch_bounds__(THREADS)
void split_embed(const float* __restrict__ embed,
                 unsigned short* __restrict__ ehg,
                 unsigned short* __restrict__ elg,
                 float* __restrict__ esqg) {
  const int wave = threadIdx.x >> 5;
  const int lane = threadIdx.x & 31;
  const int code = blockIdx.x * 8 + wave;

  const float* src = embed + (size_t)code * DIMS + lane * 8;
  union { unsigned short u[8]; uint4 q; } hb, lb;
  float ssq = 0.f;
#pragma unroll
  for (int q = 0; q < 8; ++q) {
    float f = src[q];
    ssq += f * f;
    unsigned short h = f32_bf16_rne(f);
    hb.u[q] = h;
    lb.u[q] = f32_bf16_rne(f - bf16_f32(h));
  }
  *(uint4*)(ehg + (size_t)code * DIMS + lane * 8) = hb.q;
  *(uint4*)(elg + (size_t)code * DIMS + lane * 8) = lb.q;
#pragma unroll
  for (int off = 16; off >= 1; off >>= 1) ssq += __shfl_xor(ssq, off, 32);
  if (lane == 0) esqg[code] = ssq;
}

// ---------------------------------------------------------------------------
// Async copy of one 32KB bf16 chunk (32 codes, hi+lo) global -> LDS.
// 8 x global_load_async_to_lds_b128 per thread, tracked by ASYNCcnt.
// ---------------------------------------------------------------------------
__device__ __forceinline__ void issue_chunk_async(
    unsigned lds_base, const unsigned short* __restrict__ ehg,
    const unsigned short* __restrict__ elg, int k0, int tid) {
  const unsigned short* gh = ehg + (size_t)k0 * DIMS;   // 16 KB hi slab
  const unsigned short* gl = elg + (size_t)k0 * DIMS;   // 16 KB lo slab
#pragma unroll
  for (int j = 0; j < 4; ++j) {
    unsigned off  = (unsigned)(tid * 16 + j * 4096);
    unsigned ldsh = lds_base + off;            // hi at +0
    unsigned ldsl = lds_base + 16384u + off;   // lo at +16KB
    asm volatile("global_load_async_to_lds_b128 %0, %1, %2 offset:0"
                 :: "v"(ldsh), "v"(off), "s"(gh) : "memory");
    asm volatile("global_load_async_to_lds_b128 %0, %1, %2 offset:0"
                 :: "v"(ldsl), "v"(off), "s"(gl) : "memory");
  }
}

// ---------------------------------------------------------------------------
// Fast path: 8 waves, each owns 16 rows x all 32 codes of the chunk.
// A fragments (x, bf16 hi+lo) pinned in registers for the whole K loop:
// steady state = 64 ds_load_b128 per 48 WMMAs + async embed stream.
// ---------------------------------------------------------------------------
__global__ __launch_bounds__(THREADS)
void vq_fast(const float* __restrict__ x,
             const float* __restrict__ embed,
             const unsigned short* __restrict__ ehg,
             const unsigned short* __restrict__ elg,
             const float* __restrict__ esqg,
             float* __restrict__ out) {
  __shared__ unsigned short ebuf[2][2 * CHUNK * DIMS];  // 2 x 32 KB
  __shared__ int fidx[ROWS_FAST];

  const int tid  = threadIdx.x;
  const int lane = tid & 31;
  const int wave = tid >> 5;         // row strip 0..7 (16 rows each)
  const int m    = lane & 15;
  const int hlf  = lane >> 4;
  const int row0 = blockIdx.x * ROWS_FAST;

  // kick off the async embed stream first so it overlaps x conversion
  unsigned ldsb0 = (unsigned)(size_t)(void*)&ebuf[0][0];
  unsigned ldsb1 = (unsigned)(size_t)(void*)&ebuf[1][0];
  issue_chunk_async(ldsb0, ehg, elg, 0 * CHUNK, tid);
  issue_chunk_async(ldsb1, ehg, elg, 1 * CHUNK, tid);

  // ---- build per-wave A fragments in registers: 16x32 bf16 layout ----
  // lane(m,hlf) of A holds, per 32-dim step s: dims s*32+hlf*16+{0..7} (h[0])
  // and s*32+hlf*16+8+{0..7} (h[1]) of row (wave*16+m).
  AB16 Ah[8], Al[8];
  {
    const float* xrow = x + (size_t)(row0 + wave * 16 + m) * DIMS + hlf * 16;
#pragma unroll
    for (int s = 0; s < 8; ++s) {
#pragma unroll
      for (int g = 0; g < 2; ++g) {               // +0 / +8 dim sub-blocks
        const float* p = xrow + s * 32 + g * 8;
        v4f f0 = *(const v4f*)p;
        v4f f1 = *(const v4f*)(p + 4);
#pragma unroll
        for (int q = 0; q < 4; ++q) {
          unsigned short h0 = f32_bf16_rne(f0[q]);
          Ah[s].us[g * 8 + q] = h0;
          Al[s].us[g * 8 + q] = f32_bf16_rne(f0[q] - bf16_f32(h0));
          unsigned short h1 = f32_bf16_rne(f1[q]);
          Ah[s].us[g * 8 + 4 + q] = h1;
          Al[s].us[g * 8 + 4 + q] = f32_bf16_rne(f1[q] - bf16_f32(h1));
        }
      }
    }
  }

  float bs[8];
  int   bi[8];
#pragma unroll
  for (int r = 0; r < 8; ++r) { bs[r] = -3.4e38f; bi[r] = 0; }

  const int brow0 = (m)      * DIMS + hlf * 16;   // B frag, codes 0..15
  const int brow1 = (16 + m) * DIMS + hlf * 16;   // B frag, codes 16..31

  for (int ci = 0; ci < NCHUNKS; ++ci) {
    // wait for this chunk's own async loads (in-order), then block-sync
    if (ci + 1 < NCHUNKS) asm volatile("s_wait_asynccnt 0x8" ::: "memory");
    else                  asm volatile("s_wait_asynccnt 0x0" ::: "memory");
    __syncthreads();

    const unsigned short* eb = &ebuf[ci & 1][0];
    v8f acc0 = {}, acc1 = {};
#pragma unroll
    for (int s = 0; s < 8; ++s) {
      const int d0 = s * 32;
      AB16 b0h, b0l, b1h, b1l;
      b0h.h[0] = *(const v8bf*)&eb[brow0 + d0];
      b0h.h[1] = *(const v8bf*)&eb[brow0 + d0 + 8];
      b1h.h[0] = *(const v8bf*)&eb[brow1 + d0];
      b1h.h[1] = *(const v8bf*)&eb[brow1 + d0 + 8];
      b0l.h[0] = *(const v8bf*)&eb[CHUNK * DIMS + brow0 + d0];
      b0l.h[1] = *(const v8bf*)&eb[CHUNK * DIMS + brow0 + d0 + 8];
      b1l.h[0] = *(const v8bf*)&eb[CHUNK * DIMS + brow1 + d0];
      b1l.h[1] = *(const v8bf*)&eb[CHUNK * DIMS + brow1 + d0 + 8];
      acc0 = __builtin_amdgcn_wmma_f32_16x16x32_bf16(false, Ah[s].v, false, b0h.v,
                                                     (short)0, acc0, false, false);
      acc0 = __builtin_amdgcn_wmma_f32_16x16x32_bf16(false, Ah[s].v, false, b0l.v,
                                                     (short)0, acc0, false, false);
      acc0 = __builtin_amdgcn_wmma_f32_16x16x32_bf16(false, Al[s].v, false, b0h.v,
                                                     (short)0, acc0, false, false);
      acc1 = __builtin_amdgcn_wmma_f32_16x16x32_bf16(false, Ah[s].v, false, b1h.v,
                                                     (short)0, acc1, false, false);
      acc1 = __builtin_amdgcn_wmma_f32_16x16x32_bf16(false, Ah[s].v, false, b1l.v,
                                                     (short)0, acc1, false, false);
      acc1 = __builtin_amdgcn_wmma_f32_16x16x32_bf16(false, Al[s].v, false, b1h.v,
                                                     (short)0, acc1, false, false);
    }

    // ---- score = 2*(x.e) - |e|^2 ; running argmax (c0 < c1, strict >) ----
    const int k0 = ci * CHUNK;
    const float e20 = esqg[k0 + m];
    const float e21 = esqg[k0 + 16 + m];
    const int c0 = k0 + m, c1 = k0 + 16 + m;
#pragma unroll
    for (int r = 0; r < 8; ++r) {
      float s0 = 2.f * acc0[r] - e20;
      float s1 = 2.f * acc1[r] - e21;
      if (s0 > bs[r]) { bs[r] = s0; bi[r] = c0; }
      if (s1 > bs[r]) { bs[r] = s1; bi[r] = c1; }
    }

    __syncthreads();   // all waves done reading ebuf[ci&1] before refill
    if (ci + 2 < NCHUNKS)
      issue_chunk_async((ci & 1) ? ldsb1 : ldsb0, ehg, elg, (ci + 2) * CHUNK, tid);
  }

  // ---- wave-local argmax: reduce across 16 lanes of each half ----
#pragma unroll
  for (int r = 0; r < 8; ++r) {
    float s = bs[r];
    int   i = bi[r];
#pragma unroll
    for (int off = 8; off >= 1; off >>= 1) {   // xor masks stay within 16-lane half
      float so = __shfl_xor(s, off, 32);
      int   io = __shfl_xor(i, off, 32);
      if (so > s || (so == s && io < i)) { s = so; i = io; }
    }
    if (m == 0)                                 // C layout: VGPR r -> M=r / M=r+8
      fidx[wave * 16 + hlf * 8 + r] = i;
  }
  __syncthreads();

  // ---- gather winning codebook rows (tid == dim, fully coalesced) ----
  for (int r = 0; r < ROWS_FAST; ++r) {
    int code = fidx[r];
    out[(size_t)(row0 + r) * DIMS + tid] = embed[(size_t)code * DIMS + tid];
  }
}

// ---------------------------------------------------------------------------
// Fallback (ws too small): self-contained, per-chunk in-kernel split.
// ---------------------------------------------------------------------------
__global__ __launch_bounds__(THREADS)
void vq_fallback(const float* __restrict__ x,
                 const float* __restrict__ embed,
                 float* __restrict__ out) {
  __shared__ unsigned short xh[64 * DIMS];
  __shared__ unsigned short xl[64 * DIMS];
  __shared__ unsigned short eh[CHUNK * DIMS];
  __shared__ unsigned short el[CHUNK * DIMS];
  __shared__ float esq_p[CHUNK][8];
  __shared__ float esq[CHUNK];
  __shared__ float red_s[64 * 2];
  __shared__ int   red_i[64 * 2];
  __shared__ int   fidx[64];

  const int tid  = threadIdx.x;
  const int lane = tid & 31;
  const int wave = tid >> 5;
  const int wrow = wave >> 1;
  const int wcol = wave & 1;
  const int row0 = blockIdx.x * 64;
  const int m    = lane & 15;
  const int hlf  = lane >> 4;

  for (int i = tid; i < 64 * DIMS; i += THREADS) {
    float v = x[(size_t)row0 * DIMS + i];
    unsigned short h = f32_bf16_rne(v);
    xh[i] = h;
    xl[i] = f32_bf16_rne(v - bf16_f32(h));
  }
  float bs[8]; int bi[8];
#pragma unroll
  for (int r = 0; r < 8; ++r) { bs[r] = -3.4e38f; bi[r] = 0; }

  for (int k0 = 0; k0 < KCODES; k0 += CHUNK) {
    __syncthreads();
    {
      const int c  = tid >> 3;
      const int d0 = (tid & 7) * 32;
      const float* src = embed + (size_t)(k0 + c) * DIMS + d0;
      float ssq = 0.f;
#pragma unroll
      for (int j = 0; j < 32; j += 4) {
        v4f v = *(const v4f*)(src + j);
#pragma unroll
        for (int q = 0; q < 4; ++q) {
          float f = v[q];
          ssq += f * f;
          unsigned short h = f32_bf16_rne(f);
          eh[c * DIMS + d0 + j + q] = h;
          el[c * DIMS + d0 + j + q] = f32_bf16_rne(f - bf16_f32(h));
        }
      }
      esq_p[c][tid & 7] = ssq;
    }
    __syncthreads();
    if (tid < CHUNK) {
      float s = 0.f;
#pragma unroll
      for (int q = 0; q < 8; ++q) s += esq_p[tid][q];
      esq[tid] = s;
    }
    __syncthreads();

    v8f acc = {};
    const int arow = (wrow * 16 + m) * DIMS + hlf * 16;
    const int brow = (wcol * 16 + m) * DIMS + hlf * 16;
#pragma unroll
    for (int d0 = 0; d0 < DIMS; d0 += 32) {
      AB16 ah, al, bh, bl;
      ah.h[0] = *(const v8bf*)&xh[arow + d0];
      ah.h[1] = *(const v8bf*)&xh[arow + d0 + 8];
      al.h[0] = *(const v8bf*)&xl[arow + d0];
      al.h[1] = *(const v8bf*)&xl[arow + d0 + 8];
      bh.h[0] = *(const v8bf*)&eh[brow + d0];
      bh.h[1] = *(const v8bf*)&eh[brow + d0 + 8];
      bl.h[0] = *(const v8bf*)&el[brow + d0];
      bl.h[1] = *(const v8bf*)&el[brow + d0 + 8];
      acc = __builtin_amdgcn_wmma_f32_16x16x32_bf16(false, ah.v, false, bh.v,
                                                    (short)0, acc, false, false);
      acc = __builtin_amdgcn_wmma_f32_16x16x32_bf16(false, ah.v, false, bl.v,
                                                    (short)0, acc, false, false);
      acc = __builtin_amdgcn_wmma_f32_16x16x32_bf16(false, al.v, false, bh.v,
                                                    (short)0, acc, false, false);
    }
    const float e2  = esq[wcol * 16 + m];
    const int  code = k0 + wcol * 16 + m;
#pragma unroll
    for (int r = 0; r < 8; ++r) {
      float s = 2.f * acc[r] - e2;
      if (s > bs[r]) { bs[r] = s; bi[r] = code; }
    }
  }

#pragma unroll
  for (int r = 0; r < 8; ++r) {
    float s = bs[r]; int i = bi[r];
#pragma unroll
    for (int off = 8; off >= 1; off >>= 1) {
      float so = __shfl_xor(s, off, 32);
      int   io = __shfl_xor(i, off, 32);
      if (so > s || (so == s && io < i)) { s = so; i = io; }
    }
    if (m == 0) {
      int row_local = wrow * 16 + hlf * 8 + r;
      red_s[row_local * 2 + wcol] = s;
      red_i[row_local * 2 + wcol] = i;
    }
  }
  __syncthreads();
  if (tid < 64) {
    float s0 = red_s[tid * 2], s1 = red_s[tid * 2 + 1];
    int   i0 = red_i[tid * 2], i1 = red_i[tid * 2 + 1];
    fidx[tid] = (s1 > s0 || (s1 == s0 && i1 < i0)) ? i1 : i0;
  }
  __syncthreads();
  for (int r = 0; r < 64; ++r) {
    int code = fidx[r];
    out[(size_t)(row0 + r) * DIMS + tid] = embed[(size_t)code * DIMS + tid];
  }
}

extern "C" void kernel_launch(void* const* d_in, const int* in_sizes, int n_in,
                              void* d_out, int out_size, void* d_ws, size_t ws_size,
                              hipStream_t stream) {
  (void)n_in; (void)out_size;
  const float* x     = (const float*)d_in[0];
  const float* embed = (const float*)d_in[1];
  float*       out   = (float*)d_out;
  const int nrows = in_sizes[0] / DIMS;   // 32768

  const size_t need = (size_t)KCODES * DIMS * 2 * 2 + (size_t)KCODES * 4; // 8MB + 32KB
  if (d_ws != nullptr && ws_size >= need) {
    unsigned short* ehg  = (unsigned short*)d_ws;
    unsigned short* elg  = ehg + (size_t)KCODES * DIMS;
    float*          esqg = (float*)(elg + (size_t)KCODES * DIMS);
    split_embed<<<dim3(KCODES / 8), dim3(THREADS), 0, stream>>>(embed, ehg, elg, esqg);
    vq_fast<<<dim3(nrows / ROWS_FAST), dim3(THREADS), 0, stream>>>(
        x, embed, ehg, elg, esqg, out);
  } else {
    vq_fallback<<<dim3(nrows / 64), dim3(THREADS), 0, stream>>>(x, embed, out);
  }
}
